// Atten_des_1563368096547
// MI455X (gfx1250) — matrix operand order repaired and checked by reading, not compiled, not run on hardware
//
#include <hip/hip_runtime.h>
#include <math.h>

// Shapes from the reference
#define B_  64
#define K_  49
#define T_  512
#define H_  512
#define A_  49
#define AP_ 64   // A (and K) padded to 64 for 16x16 WMMA tiling

typedef __attribute__((ext_vector_type(16))) _Float16 v16h;
typedef __attribute__((ext_vector_type(8)))  float    v8f;

__device__ __forceinline__ float warpMax(float v) {
#pragma unroll
  for (int off = 16; off > 0; off >>= 1) v = fmaxf(v, __shfl_xor(v, off, 32));
  return v;
}
__device__ __forceinline__ float warpSum(float v) {
#pragma unroll
  for (int off = 16; off > 0; off >>= 1) v += __shfl_xor(v, off, 32);
  return v;
}

// Hardware tanh (CDNA5 V_TANH_F32 trans op). Two independent tanh back-to-back:
// the 2nd is the "1 independent op" required after the 1st (ISA 5.7.4), and the
// trailing v_nop covers the 2nd before its result is consumed. Early-clobber
// outputs so regalloc can't alias an output register onto the 2nd input.
__device__ __forceinline__ void tanh2_hw(float x1, float x2, float& y1, float& y2) {
  asm("v_tanh_f32 %0, %2\n\t"
      "v_tanh_f32 %1, %3\n\t"
      "v_nop"
      : "=&v"(y1), "=&v"(y2)
      : "v"(x1), "v"(x2));
}
__device__ __forceinline__ float tanh1_hw(float x) {
  float y;
  asm("v_tanh_f32 %0, %1\n\t"
      "v_nop"
      : "=v"(y)
      : "v"(x));
  return y;
}

// ---------------------------------------------------------------------------
// Pre-pass A: pack Wv|Wg|Ws (49x512 f32) into f16, zero-padded to 64x512.
// ---------------------------------------------------------------------------
__global__ __launch_bounds__(256) void pad_w_kernel(const float* __restrict__ Wv,
                                                    const float* __restrict__ Wg,
                                                    const float* __restrict__ Ws,
                                                    _Float16* __restrict__ Wp) {
  const int idx = blockIdx.x * 256 + threadIdx.x;   // over 3*64*512
  const int w = idx / (AP_ * H_);
  const int rem = idx - w * (AP_ * H_);
  const int a = rem / H_;
  const int h = rem - a * H_;
  const float* src = (w == 0) ? Wv : (w == 1) ? Wg : Ws;
  Wp[idx] = (a < A_) ? (_Float16)src[a * H_ + h] : (_Float16)0.f;
}

// ---------------------------------------------------------------------------
// Pre-pass B: VT16[b][h][k] = (f16) V[b][k][h], k zero-padded to 64.
// LDS tile transpose so both global read and write are coalesced.
// grid (B, H/64), block 256
// ---------------------------------------------------------------------------
__global__ __launch_bounds__(256) void vt_kernel(const float* __restrict__ V,
                                                 _Float16* __restrict__ VT) {
  __shared__ _Float16 tile[64 * 74];   // [k][h], stride 74 halves: conflict-free
  const int tid = threadIdx.x;
  const int b = blockIdx.x;
  const int h0 = blockIdx.y * 64;
  const float* Vb = V + (size_t)b * K_ * H_;
  for (int i = tid; i < 64 * 64; i += 256) {
    const int k = i >> 6, h = i & 63;
    tile[k * 74 + h] = (k < K_) ? (_Float16)Vb[(size_t)k * H_ + h0 + h] : (_Float16)0.f;
  }
  __syncthreads();
  for (int i = tid; i < 64 * 64; i += 256) {
    const int h = i >> 6, k = i & 63;     // k innermost -> coalesced stores
    VT[((size_t)b * H_ + h0 + h) * AP_ + k] = tile[k * 74 + h];
  }
}

// ---------------------------------------------------------------------------
// Kernel 1: Out[r, a] = sum_h X[r,h] * W[a,h], a padded to 64 (pad cols = 0).
// One wave -> 16 rows x 64 cols; 16 steps of v_wmma_f32_16x16x32_f16.
// R is always a multiple of 16, so surviving waves are fully in range:
// no load clamps, no store guards -> straight-line epilogue.
// ---------------------------------------------------------------------------
__global__ __launch_bounds__(256) void proj_kernel(const float* __restrict__ X,
                                                   const _Float16* __restrict__ Wp,
                                                   float* __restrict__ Out, int R) {
  const int lane = threadIdx.x & 31;
  const int wave = threadIdx.x >> 5;
  const int r0 = (blockIdx.x * 8 + wave) * 16;
  if (r0 >= R) return;                 // wave-uniform; no barriers in this kernel
  const int hh = lane >> 4;            // 16-lane half select
  const int ln = lane & 15;
  const float* xrow = X + (size_t)(r0 + ln) * H_;

  v8f acc[4] = {v8f{}, v8f{}, v8f{}, v8f{}};

#pragma unroll 2
  for (int h0 = 0; h0 < H_; h0 += 32) {
    // prefetch the A stream one iteration ahead (global_prefetch_b8)
    __builtin_prefetch(xrow + h0 + 32, 0, 1);
    // A fragment: af[8*half + j] <-> K = 16*hh + 8*half + j (contiguous 16 per lane half)
    v16h af;
    const float* xr = xrow + h0;
#pragma unroll
    for (int half = 0; half < 2; ++half) {
      const int base = 16 * hh + 8 * half;
#pragma unroll
      for (int j = 0; j < 8; ++j) af[8 * half + j] = (_Float16)xr[base + j];
    }
    // B fragments: bf[i] <-> K = 16*hh + i at row a = nt*16+ln : contiguous 32B
#pragma unroll
    for (int nt = 0; nt < 4; ++nt) {
      const int a = nt * 16 + ln;
      const v16h bf = *(const v16h*)(Wp + (size_t)a * H_ + h0 + 16 * hh);
      acc[nt] = __builtin_amdgcn_wmma_f32_16x16x32_f16(
          false, af, false, bf, (short)0, acc[nt], false, false);
    }
  }
  // C/D layout: VGPR j -> (M=j, N=lane) lanes 0-15, (M=j+8) lanes 16-31
#pragma unroll
  for (int nt = 0; nt < 4; ++nt) {
    const int col = nt * 16 + ln;
#pragma unroll
    for (int j = 0; j < 8; ++j) {
      const int row = r0 + j + 8 * hh;
      Out[(size_t)row * AP_ + col] = acc[nt][j];
    }
  }
}

// ---------------------------------------------------------------------------
// Kernel 2: per (b,t) additive-attention scores + softmax + beta.
// One wave per t; cv[b] (49x65-strided) + wh in LDS. Hardware v_tanh_f32 for
// the 79M-element tanh term (the pipeline's compute bottleneck). Emits f32
// alpha (output), padded f16 alpha row (for the WMMA context kernel), beta.
// ---------------------------------------------------------------------------
__global__ __launch_bounds__(256) void score_kernel(const float* __restrict__ cvWs,
                                                    const float* __restrict__ ghWs,
                                                    const float* __restrict__ svWs,
                                                    const float* __restrict__ Wh,
                                                    float* __restrict__ alphaOut,
                                                    _Float16* __restrict__ alphaPad,
                                                    float* __restrict__ betaOut) {
  __shared__ float cvS[K_ * 65];
  __shared__ float whS[64];
  __shared__ float ghS[8 * 64];

  const int tid = threadIdx.x;
  const int lane = tid & 31, wave = tid >> 5;
  const int b = blockIdx.x;
  const int t = blockIdx.y * 8 + wave;
  const int bt = b * T_ + t;

  for (int idx = tid; idx < K_ * AP_; idx += 256) {
    const int k = idx >> 6, a = idx & 63;
    cvS[k * 65 + a] = cvWs[(size_t)(b * K_ + k) * AP_ + a];
  }
  if (tid < 64) whS[tid] = (tid < A_) ? Wh[tid] : 0.f;
  ghS[wave * 64 + lane]      = ghWs[(size_t)bt * AP_ + lane];
  ghS[wave * 64 + lane + 32] = ghWs[(size_t)bt * AP_ + lane + 32];
  __syncthreads();

  const float* ghR = &ghS[wave * 64];
  const int k1 = lane;                 // always < 49
  const int k2 = lane + 32;
  const bool v2 = (k2 < K_);
  const int k2c = v2 ? k2 : 0;

  float z1 = 0.f, z2 = 0.f;
#pragma unroll 7
  for (int a = 0; a < A_; ++a) {
    const float g = ghR[a], w = whS[a];
    float t1, t2;
    tanh2_hw(cvS[k1 * 65 + a] + g, cvS[k2c * 65 + a] + g, t1, t2);
    z1 += t1 * w;
    z2 += t2 * w;
  }
  const float m  = warpMax(v2 ? fmaxf(z1, z2) : z1);
  const float e1 = __expf(z1 - m);
  const float e2 = v2 ? __expf(z2 - m) : 0.f;
  const float S  = warpSum(e1 + e2);
  const float inv = 1.f / S;
  const float a1 = e1 * inv;
  const float a2 = e2 * inv;            // exact 0 for invalid lanes
  alphaOut[(size_t)bt * K_ + k1] = a1;
  if (v2) alphaOut[(size_t)bt * K_ + k2] = a2;
  alphaPad[(size_t)bt * AP_ + k1] = (_Float16)a1;   // lanes cover 0..31
  alphaPad[(size_t)bt * AP_ + k2] = (_Float16)a2;   // and 32..63 (pad = 0)

  // extended score z_ext from s_t projection; softmax over [z_t, z_ext] -> beta
  const float* svR = svWs + (size_t)bt * AP_;
  float p = tanh1_hw(svR[k1] + ghR[k1]) * whS[k1];
  if (v2) p += tanh1_hw(svR[k2] + ghR[k2]) * whS[k2];
  const float ze = warpSum(p);
  const float m2 = fmaxf(m, ze);
  const float S2 = S * __expf(m - m2) + __expf(ze - m2);
  if (lane == 0) betaOut[bt] = __expf(ze - m2) / S2;
}

// ---------------------------------------------------------------------------
// Kernel 3: c_t = alpha @ V[b] via WMMA (K padded to 64, two 32-k steps),
// fused epilogue c_hat = beta*s_t + (1-beta)*c_t.
// A from padded f16 alpha, B from pre-transposed f16 V: all loads contiguous.
// ---------------------------------------------------------------------------
__global__ __launch_bounds__(256) void ctx_kernel(const _Float16* __restrict__ alphaPad,
                                                  const float* __restrict__ betaOut,
                                                  const _Float16* __restrict__ VT,
                                                  const float* __restrict__ s_t,
                                                  float* __restrict__ cHat) {
  const int lane = threadIdx.x & 31, wave = threadIdx.x >> 5;
  const int bt0  = blockIdx.x * 16;            // row tile never crosses a batch
  const int b    = bt0 / T_;
  const int col0 = (blockIdx.y * 8 + wave) * 16;
  const int hh = lane >> 4, ln = lane & 15;

  const _Float16* ar = alphaPad + (size_t)(bt0 + ln) * AP_;
  const _Float16* vr = VT + ((size_t)b * H_ + col0 + ln) * AP_;

  // lanes 0..15 fetch beta for the 16 rows of this tile; rows picked by shuffle
  const float betaLane = betaOut[bt0 + (lane & 15)];

  v8f acc = {};
#pragma unroll
  for (int k0 = 0; k0 < 64; k0 += 32) {
    v16h af;
#pragma unroll
    for (int half = 0; half < 2; ++half) {
      const int base = k0 + 16 * hh + 8 * half;
#pragma unroll
      for (int j = 0; j < 8; ++j) af[8 * half + j] = ar[base + j];
    }
    const v16h bf = *(const v16h*)(vr + k0 + 16 * hh);
    acc = __builtin_amdgcn_wmma_f32_16x16x32_f16(
        false, af, false, bf, (short)0, acc, false, false);
  }
#pragma unroll
  for (int j = 0; j < 8; ++j) {
    const int m = j + 8 * hh;                  // row within tile
    const int bt = bt0 + m;
    const int col = col0 + ln;
    const float beta = __shfl(betaLane, m, 32);
    const float sv = s_t[(size_t)bt * H_ + col];
    cHat[(size_t)bt * H_ + col] = beta * sv + (1.f - beta) * acc[j];
  }
}

// ---------------------------------------------------------------------------
extern "C" void kernel_launch(void* const* d_in, const int* in_sizes, int n_in,
                              void* d_out, int out_size, void* d_ws, size_t ws_size,
                              hipStream_t stream) {
  const float* V   = (const float*)d_in[0];
  const float* h_t = (const float*)d_in[1];
  const float* s_t = (const float*)d_in[2];
  const float* Wv  = (const float*)d_in[3];
  const float* Wg  = (const float*)d_in[4];
  const float* Ws  = (const float*)d_in[5];
  const float* Wh  = (const float*)d_in[6];

  // workspace layout:
  //   f32:  gh (B*T*64) | sv (B*T*64) | cv (B*K*64)         = 17.6 MB
  //   f16:  Wpad x3 (3*64*512) | VT (B*512*64) | alphaPad (B*T*64) = 8.6 MB
  float* ws = (float*)d_ws;
  const size_t ghN = (size_t)B_ * T_ * AP_;      // 2,097,152
  const size_t cvN = (size_t)B_ * K_ * AP_;      // 200,704
  float* gh = ws;
  float* sv = ws + ghN;
  float* cv = ws + 2 * ghN;
  _Float16* h16 = (_Float16*)(ws + 2 * ghN + cvN);
  _Float16* Wp      = h16;                                   // 3*64*512
  _Float16* VT      = h16 + (size_t)3 * AP_ * H_;            // B*512*64
  _Float16* alphaPad= VT + (size_t)B_ * H_ * AP_;            // B*T*64

  float* cHat     = (float*)d_out;
  float* alphaOut = cHat + (size_t)B_ * T_ * H_;
  float* betaOut  = alphaOut + (size_t)B_ * T_ * K_;

  const int RT = B_ * T_;   // 32768 rows
  const int RK = B_ * K_;   // 3136 rows (multiple of 16)

  pad_w_kernel<<<dim3(3 * AP_ * H_ / 256), 256, 0, stream>>>(Wv, Wg, Ws, Wp);
  vt_kernel<<<dim3(B_, H_ / 64), 256, 0, stream>>>(V, VT);
  proj_kernel<<<dim3((RT + 127) / 128), 256, 0, stream>>>(h_t, Wp + (size_t)1 * AP_ * H_, gh, RT);
  proj_kernel<<<dim3((RT + 127) / 128), 256, 0, stream>>>(s_t, Wp + (size_t)2 * AP_ * H_, sv, RT);
  proj_kernel<<<dim3((RK + 127) / 128), 256, 0, stream>>>(V, Wp, cv, RK);
  score_kernel<<<dim3(B_, T_ / 8), 256, 0, stream>>>(cv, gh, sv, Wh, alphaOut, alphaPad, betaOut);
  ctx_kernel<<<dim3(RT / 16, 4), 256, 0, stream>>>(alphaPad, betaOut, VT, s_t, cHat);
}
